// DomainAdaption_60756607369486
// MI455X (gfx1250) — compile-verified, matrix operand order checked by reading, not run on hardware
//
#include <hip/hip_runtime.h>
#include <hip/hip_bf16.h>
#include <math.h>

typedef _Float16 v16h __attribute__((ext_vector_type(16)));
typedef _Float16 v8h  __attribute__((ext_vector_type(8)));
typedef float    v8f  __attribute__((ext_vector_type(8)));

#define Nn 16
#define Cc 128
#define Hh 128
#define Ww 128
#define CH 32            // adapter hidden = C/4

// LDS staging tile: 3 rows x 66 cols x 128 channels (f16), channel-contiguous,
// padded to 136 halves per (row,col) so ds_load_b128 stays 16B-aligned and
// bank conflicts are limited to 2-way.
#define LDS_CSTRIDE 136
#define LDS_PAIRS   (3*66)

// ---------------------------------------------------------------------------
// Pack conv weights [Cout,Cin,3,3] f32 into per-lane WMMA A-fragments (f16).
// Fragment index: (((mtile*9 + tap)*4 + kchunk)*32 + lane)*16 + j
// K mapping per ISA 16-bit A layout: k = (lane>=16 ? 8:0) + (j<8 ? j : j+8)
// ---------------------------------------------------------------------------
__global__ void pack_wmma_weights(const float* __restrict__ w, _Float16* __restrict__ frag) {
    int idx = blockIdx.x * blockDim.x + threadIdx.x;
    const int total = 8 * 9 * 4 * 32 * 16;
    if (idx >= total) return;
    int j    = idx & 15;
    int r1   = idx >> 4;
    int lane = r1 & 31;
    int r2   = r1 >> 5;
    int kc   = r2 & 3;
    int r3   = r2 >> 2;
    int t    = r3 % 9;
    int mt   = r3 / 9;
    int M    = lane & 15;
    int kb   = (lane >> 4) << 3;
    int k    = kb + (j < 8 ? j : j + 8);
    int cout = mt * 16 + M;
    int ci   = kc * 32 + k;
    frag[idx] = (_Float16)w[(cout * Cc + ci) * 9 + t];
}

// ---------------------------------------------------------------------------
// 3x3 conv via implicit GEMM with v_wmma_f32_16x16x32_f16.
// MODE 0: input f32 (x), output f16 with PReLU  -> h1
// MODE 1: input f16 (h1), output f32 no act     -> h
// Block: 256 threads (8 waves). Grid: (W/64, H, N).
// Wave partition: mpair = wave&3 -> M-tiles {2*mpair, 2*mpair+1};
//                 nhalf = wave>>2 -> N-subtiles {2*nhalf, 2*nhalf+1}.
// Each B fragment feeds 2 WMMAs (both M-tiles) -> 1 ds_load_b128 per WMMA.
// ---------------------------------------------------------------------------
template<int MODE>
__global__ __launch_bounds__(256)
void conv3x3_wmma(const void* __restrict__ in_,
                  const _Float16* __restrict__ wfrag,
                  const float* __restrict__ bias,
                  const float* __restrict__ slope_ptr,
                  _Float16* __restrict__ out16,
                  float* __restrict__ out32)
{
    __shared__ _Float16 sx[LDS_PAIRS * LDS_CSTRIDE];

    const int tid = threadIdx.x;
    const int n   = blockIdx.z;
    const int h0  = blockIdx.y;
    const int w0  = blockIdx.x * 64;

    // ---- stage input halo tile to LDS (with f32->f16 convert for MODE 0) ----
    for (int idx = tid; idx < Cc * LDS_PAIRS; idx += 256) {
        int ci = idx / LDS_PAIRS;
        int pr = idx - ci * LDS_PAIRS;          // (row,col) pair, col-contiguous
        int r  = pr / 66;
        int c  = pr - r * 66;
        int gr = h0 + r - 1;
        int gc = w0 + c - 1;
        _Float16 v = (_Float16)0.f;
        if ((unsigned)gr < (unsigned)Hh && (unsigned)gc < (unsigned)Ww) {
            size_t g = ((((size_t)n * Cc + ci) * Hh + gr) << 7) + gc;
            if (MODE == 0) v = (_Float16)((const float*)in_)[g];
            else           v = ((const _Float16*)in_)[g];
        }
        sx[pr * LDS_CSTRIDE + ci] = v;
    }
    __syncthreads();

    const int lane  = tid & 31;
    const int wave  = tid >> 5;
    const int mt0   = (wave & 3) * 2;           // first of 2 M-tiles
    const int ns0   = (wave >> 2) * 2;          // first of 2 N-subtiles
    const int nlo   = lane & 15;
    const int koff  = (lane >> 4) << 3;         // 0 or 8

    v8f acc[2][2] = {};                         // [nt][mi], 32 VGPRs total

    for (int t = 0; t < 9; ++t) {               // 3x3 taps
        int r  = t / 3;
        int kw = t - r * 3;
        #pragma unroll
        for (int kc = 0; kc < 4; ++kc) {        // 128 in-channels in K=32 chunks
            // two A fragments (adjacent M-tiles): aligned 32B vector loads
            v16h a0 = *(const v16h*)(wfrag + (((((mt0 + 0) * 9 + t) * 4 + kc) * 32 + lane) << 4));
            v16h a1 = *(const v16h*)(wfrag + (((((mt0 + 1) * 9 + t) * 4 + kc) * 32 + lane) << 4));
            #pragma unroll
            for (int nt = 0; nt < 2; ++nt) {    // 2 N-subtiles of 16 columns
                int col  = (ns0 + nt) * 16 + nlo + kw;      // 0..65
                int base = (r * 66 + col) * LDS_CSTRIDE + kc * 32 + koff;
                v8h lo = *(const v8h*)(sx + base);          // K = koff..koff+7
                v8h hi = *(const v8h*)(sx + base + 16);     // K = koff+16..koff+23
                v16h b = __builtin_shufflevector(lo, hi,
                         0,1,2,3,4,5,6,7,8,9,10,11,12,13,14,15);
                acc[nt][0] = __builtin_amdgcn_wmma_f32_16x16x32_f16(
                                 false, a0, false, b, (short)0, acc[nt][0], false, false);
                acc[nt][1] = __builtin_amdgcn_wmma_f32_16x16x32_f16(
                                 false, a1, false, b, (short)0, acc[nt][1], false, false);
            }
        }
    }

    // ---- epilogue: bias (+ PReLU for MODE 0), store ----
    float s1 = 0.f;
    if (MODE == 0) s1 = *slope_ptr;
    #pragma unroll
    for (int mi = 0; mi < 2; ++mi) {
        const int cout_base = (mt0 + mi) * 16 + koff;   // C/D layout: lane>=16 -> M+8
        #pragma unroll
        for (int v = 0; v < 8; ++v) {
            int   cout = cout_base + v;
            float bv   = bias[cout];
            #pragma unroll
            for (int nt = 0; nt < 2; ++nt) {
                int w = w0 + (ns0 + nt) * 16 + nlo;
                float y = acc[nt][mi][v] + bv;
                size_t g = ((((size_t)n * Cc + cout) * Hh + h0) << 7) + w;
                if (MODE == 0) {
                    y = (y >= 0.f) ? y : s1 * y;
                    out16[g] = (_Float16)y;
                } else {
                    out32[g] = y;
                }
            }
        }
    }
}

// ---------------------------------------------------------------------------
// Global average pool: one block per (n,c), mean over 128x128.
// ---------------------------------------------------------------------------
__global__ void pool_mean(const float* __restrict__ h, float* __restrict__ x1) {
    __shared__ float red[256];
    size_t base = (size_t)blockIdx.x * (Hh * Ww);
    float s = 0.f;
    for (int i = threadIdx.x; i < Hh * Ww; i += 256) s += h[base + i];
    red[threadIdx.x] = s;
    __syncthreads();
    for (int off = 128; off > 0; off >>= 1) {
        if (threadIdx.x < off) red[threadIdx.x] += red[threadIdx.x + off];
        __syncthreads();
    }
    if (threadIdx.x == 0) x1[blockIdx.x] = red[0] * (1.f / (Hh * Ww));
}

// ---------------------------------------------------------------------------
// Per-sample expert adapter: a = relu(W1[e] x1 + b1); g = W2[e] a + b2;
// gate = sigmoid(g). Tiny: single block.
// ---------------------------------------------------------------------------
__global__ void adapter_gate(const float* __restrict__ x1, const int* __restrict__ inten,
                             const float* __restrict__ aW1, const float* __restrict__ ab1,
                             const float* __restrict__ aW2, const float* __restrict__ ab2,
                             float* __restrict__ gate)
{
    __shared__ float a_sh[Nn * CH];
    int tid = threadIdx.x;                      // 512 threads
    {
        int n  = tid >> 5;                      // 0..15
        int hh = tid & 31;                      // 0..31
        int e  = inten[n] - 1;
        float s = ab1[e * CH + hh];
        const float* wr = aW1 + (size_t)(e * CH + hh) * Cc;
        const float* xr = x1 + n * Cc;
        for (int c = 0; c < Cc; ++c) s += xr[c] * wr[c];
        a_sh[tid] = s > 0.f ? s : 0.f;
    }
    __syncthreads();
    for (int item = tid; item < Nn * Cc; item += 512) {
        int n = item >> 7;
        int c = item & 127;
        int e = inten[n] - 1;
        float g = ab2[e * Cc + c];
        const float* wr = aW2 + (size_t)(e * Cc + c) * CH;
        const float* ar = a_sh + n * CH;
        for (int hh = 0; hh < CH; ++hh) g += ar[hh] * wr[hh];
        gate[item] = 1.f / (1.f + expf(-g));
    }
}

// ---------------------------------------------------------------------------
// out = prelu(h * gate[n,c] + x, p2) -- float4 streaming kernel.
// ---------------------------------------------------------------------------
__global__ void fuse_out(const float* __restrict__ h, const float* __restrict__ x,
                         const float* __restrict__ gate, const float* __restrict__ p2ptr,
                         float* __restrict__ out)
{
    const float p2 = *p2ptr;
    const size_t total4 = (size_t)Nn * Cc * Hh * Ww / 4;
    const float4* h4 = (const float4*)h;
    const float4* x4 = (const float4*)x;
    float4* o4 = (float4*)out;
    for (size_t i = blockIdx.x * (size_t)blockDim.x + threadIdx.x; i < total4;
         i += (size_t)gridDim.x * blockDim.x) {
        size_t nc = (i * 4) >> 14;              // /(H*W)
        float gt  = gate[nc];
        float4 hv = h4[i], xv = x4[i], y;
        y.x = hv.x * gt + xv.x;  y.x = y.x >= 0.f ? y.x : p2 * y.x;
        y.y = hv.y * gt + xv.y;  y.y = y.y >= 0.f ? y.y : p2 * y.y;
        y.z = hv.z * gt + xv.z;  y.z = y.z >= 0.f ? y.z : p2 * y.z;
        y.w = hv.w * gt + xv.w;  y.w = y.w >= 0.f ? y.w : p2 * y.w;
        o4[i] = y;
    }
}

// ---------------------------------------------------------------------------
extern "C" void kernel_launch(void* const* d_in, const int* in_sizes, int n_in,
                              void* d_out, int out_size, void* d_ws, size_t ws_size,
                              hipStream_t stream) {
    const float* x     = (const float*)d_in[0];
    const int*   inten = (const int*)  d_in[1];
    const float* c1w   = (const float*)d_in[2];
    const float* c1b   = (const float*)d_in[3];
    const float* p1    = (const float*)d_in[4];
    const float* c2w   = (const float*)d_in[5];
    const float* c2b   = (const float*)d_in[6];
    const float* aW1   = (const float*)d_in[7];
    const float* ab1   = (const float*)d_in[8];
    const float* aW2   = (const float*)d_in[9];
    const float* ab2   = (const float*)d_in[10];
    const float* p2    = (const float*)d_in[11];
    float* out = (float*)d_out;

    // workspace carve-out
    char* ws = (char*)d_ws;
    const size_t FRAG_BYTES = (size_t)8 * 9 * 4 * 32 * 16 * sizeof(_Float16); // 294912
    _Float16* fragA1 = (_Float16*)ws;
    _Float16* fragA2 = (_Float16*)(ws + FRAG_BYTES);
    _Float16* h1     = (_Float16*)(ws + 2 * FRAG_BYTES);                       // 64 MiB
    const size_t H1_BYTES = (size_t)Nn * Cc * Hh * Ww * sizeof(_Float16);
    float* hbuf = (float*)(ws + 2 * FRAG_BYTES + H1_BYTES);                    // 128 MiB
    const size_t H_BYTES = (size_t)Nn * Cc * Hh * Ww * sizeof(float);
    float* x1   = (float*)(ws + 2 * FRAG_BYTES + H1_BYTES + H_BYTES);
    float* gate = x1 + Nn * Cc;

    // 1) pre-swizzle conv weights into WMMA A-fragment layout
    pack_wmma_weights<<<576, 256, 0, stream>>>(c1w, fragA1);
    pack_wmma_weights<<<576, 256, 0, stream>>>(c2w, fragA2);

    // 2) conv1 + PReLU (f32 in -> f16 out), conv2 (f16 in -> f32 out)
    dim3 cgrid(Ww / 64, Hh, Nn);
    conv3x3_wmma<0><<<cgrid, 256, 0, stream>>>(x,  fragA1, c1b, p1,      h1, nullptr);
    conv3x3_wmma<1><<<cgrid, 256, 0, stream>>>(h1, fragA2, c2b, nullptr, nullptr, hbuf);

    // 3) global average pool -> x1[n,c]
    pool_mean<<<Nn * Cc, 256, 0, stream>>>(hbuf, x1);

    // 4) per-sample expert adapter -> sigmoid gate[n,c]
    adapter_gate<<<1, 512, 0, stream>>>(x1, inten, aW1, ab1, aW2, ab2, gate);

    // 5) gated residual + PReLU
    fuse_out<<<8192, 256, 0, stream>>>(hbuf, x, gate, p2, out);
}
